// GATMultiTaskClsReg_89550068122061
// MI455X (gfx1250) — matrix operand order repaired and checked by reading, not compiled, not run on hardware
//
#include <hip/hip_runtime.h>
#include <hip/hip_bf16.h>
#include <cstdint>

typedef __attribute__((ext_vector_type(16))) _Float16 v16h;
typedef __attribute__((ext_vector_type(8)))  _Float16 v8h;
typedef __attribute__((ext_vector_type(8)))  float    v8f;

#define NEG_SLOPE 0.2f

// ---- monotone float <-> uint mapping so unsigned atomicMax == float max ----
static __device__ __forceinline__ unsigned f2u_ord(float f) {
  unsigned b = __float_as_uint(f);
  return (b & 0x80000000u) ? ~b : (b | 0x80000000u);
}
static __device__ __forceinline__ float u2f_ord(unsigned u) {
  unsigned b = (u & 0x80000000u) ? (u & 0x7fffffffu) : ~u;
  return __uint_as_float(b);
}

// one-time f32 -> f16 conversion (weights once per launch; x once)
__global__ void cvt_f32_f16(const float* __restrict__ src,
                            _Float16* __restrict__ dst, int n) {
  int i = blockIdx.x * blockDim.x + threadIdx.x;
  if (i < n) dst[i] = (_Float16)src[i];
}

// ============================================================================
// GEMM: Hout[M, NC] = A[M, K] @ W^T   (A, W pre-converted to f16; W is [NC,K])
// Block = 256 thr = 8 waves -> 128(M) x 64(N) tile.
// Double-buffered LDS B tile: while the 4 WMMAs of step k consume buffer b,
// the async fill (global_load_async_to_lds_b128) of step k+1 streams into
// buffer b^1; only s_wait_asynccnt 0x1 (in-order completion) gates the barrier.
// ============================================================================
__global__ void gemm_xwt_wmma(const _Float16* __restrict__ Ah,
                              const _Float16* __restrict__ Bh,
                              float* __restrict__ Hout,
                              int M, int K, int NC) {
  __shared__ __align__(128) _Float16 Bs[2][64 * 32];   // 2 x 4 KB B tiles

  const int lane = threadIdx.x & 31;
  const int wave = threadIdx.x >> 5;
  const int half = lane >> 4;       // 0 -> lanes 0-15, 1 -> lanes 16-31
  const int lm   = lane & 15;
  const int m0 = blockIdx.x * 128 + wave * 16;
  const int n0 = blockIdx.y * 64;

  // clamp tail rows: unconditional loads, guarded stores (no EXEC divergence)
  const int arow  = m0 + lm;
  const int arowc = (arow < M) ? arow : (M - 1);
  const _Float16* __restrict__ Arow = Ah + (size_t)arowc * K;

  // staging assignment: 256 threads move 256 x 16B chunks = 64 cols x 32 halfs
  const int scol = threadIdx.x >> 2;       // 0..63
  const int schk = threadIdx.x & 3;        // 0..3 (8 halfs each)
  const _Float16* sgp = Bh + (size_t)(n0 + scol) * K + schk * 8;
  const unsigned sldsBase =
      (unsigned)(uintptr_t)(&Bs[0][0]) + (unsigned)(scol * 32 + schk * 8) * 2u;

  auto stage = [&](int kb, int buf) {
    unsigned long long ga = (unsigned long long)(uintptr_t)(sgp + kb);
    unsigned lds = sldsBase + (unsigned)buf * (64u * 32u * 2u);
    asm volatile("global_load_async_to_lds_b128 %0, %1, off"
                 :: "v"(lds), "v"(ga) : "memory");
  };

  v8f acc0 = {}, acc1 = {}, acc2 = {}, acc3 = {};

  stage(0, 0);                       // prologue fill of buffer 0
  int buf = 0;
  for (int kb = 0; kb < K; kb += 32) {
    // kick off next tile's fill into the other buffer (wraps harmlessly at end)
    int kn = (kb + 32 < K) ? (kb + 32) : 0;
    stage(kn, buf ^ 1);
    // oldest in-flight async fill (current buffer) must be done; leave 1 pending
    asm volatile("s_wait_asynccnt 0x1" ::: "memory");
    __syncthreads();

    // A fragment: contiguous k order (half h: K h*16..+15), matching B
    v8h lo = *(const v8h*)(Arow + kb + half * 16);
    v8h hi = *(const v8h*)(Arow + kb + half * 16 + 8);
    v16h a;
    #pragma unroll
    for (int i = 0; i < 8; ++i) { a[i] = lo[i]; a[8 + i] = hi[i]; }

    // 4 WMMAs reusing the A fragment; B from LDS (ds_load_b128 x2 each)
    const _Float16* Bsb = &Bs[buf][0];
    v16h b0 = *(const v16h*)(Bsb + (0 * 16 + lm) * 32 + half * 16);
    acc0 = __builtin_amdgcn_wmma_f32_16x16x32_f16(false, a, false, b0,
                                                  (short)0, acc0, false, false);
    v16h b1 = *(const v16h*)(Bsb + (1 * 16 + lm) * 32 + half * 16);
    acc1 = __builtin_amdgcn_wmma_f32_16x16x32_f16(false, a, false, b1,
                                                  (short)0, acc1, false, false);
    v16h b2 = *(const v16h*)(Bsb + (2 * 16 + lm) * 32 + half * 16);
    acc2 = __builtin_amdgcn_wmma_f32_16x16x32_f16(false, a, false, b2,
                                                  (short)0, acc2, false, false);
    v16h b3 = *(const v16h*)(Bsb + (3 * 16 + lm) * 32 + half * 16);
    acc3 = __builtin_amdgcn_wmma_f32_16x16x32_f16(false, a, false, b3,
                                                  (short)0, acc3, false, false);

    __syncthreads();   // all reads of buf done before it is refilled next step
    buf ^= 1;
  }

  // C/D layout: VGPR r -> row (r + half*8), col = lm
  #pragma unroll
  for (int r = 0; r < 8; ++r) {
    int row = m0 + r + half * 8;
    if (row < M) {
      float* o = Hout + (size_t)row * NC + n0 + lm;
      o[0]  = acc0[r];
      o[16] = acc1[r];
      o[32] = acc2[r];
      o[48] = acc3[r];
    }
  }
}

// ============================================================================
// alo[n,h] = sum_c h[n,h,c]*a_src[h,c] ; ahi likewise with a_dst
// ============================================================================
__global__ void attn_coef(const float* __restrict__ Hf,
                          const float* __restrict__ a_src,
                          const float* __restrict__ a_dst,
                          float* __restrict__ alo, float* __restrict__ ahi,
                          int N, int H, int C) {
  int i = blockIdx.x * blockDim.x + threadIdx.x;
  if (i >= N * H) return;
  int n = i / H, h = i % H;
  const float* hp = Hf + (size_t)n * H * C + (size_t)h * C;
  const float* as = a_src + h * C;
  const float* ad = a_dst + h * C;
  float slo = 0.f, shi = 0.f;
  for (int c = 0; c < C; ++c) {
    float v = hp[c];
    slo += v * as[c];
    shi += v * ad[c];
  }
  alo[i] = slo;
  ahi[i] = shi;
}

// zero max/denominator state and the output accumulator
__global__ void layer_init(unsigned* __restrict__ mx, float* __restrict__ den,
                           float* __restrict__ out, int NH, int NHC) {
  int i = blockIdx.x * blockDim.x + threadIdx.x;
  if (i < NHC) out[i] = 0.0f;
  if (i < NH) { mx[i] = 0u; den[i] = 0.0f; }
}

// phase 1: e = leakyrelu(alo[src]+ahi[dst]); store e; segment-max over dst
__global__ void edge_max(const int* __restrict__ ei,
                         const float* __restrict__ alo,
                         const float* __restrict__ ahi,
                         float* __restrict__ eb, unsigned* __restrict__ mx,
                         int E, int Ep, int H) {
  int e = blockIdx.x * blockDim.x + threadIdx.x;
  if (e >= Ep) return;
  int s, d;
  if (e < E) { s = ei[e]; d = ei[E + e]; } else { s = e - E; d = s; }
  for (int h = 0; h < H; ++h) {
    float v = alo[s * H + h] + ahi[d * H + h];
    v = (v > 0.0f) ? v : NEG_SLOPE * v;
    eb[(size_t)e * H + h] = v;
    atomicMax(&mx[d * H + h], f2u_ord(v));
  }
}

// phase 2: ex = exp(e - max[dst]); store ex; segment-sum denominator
__global__ void edge_exp(const int* __restrict__ ei, float* __restrict__ eb,
                         const unsigned* __restrict__ mx,
                         float* __restrict__ den, int E, int Ep, int H) {
  int e = blockIdx.x * blockDim.x + threadIdx.x;
  if (e >= Ep) return;
  int d;
  if (e < E) { d = ei[E + e]; } else { d = e - E; }
  for (int h = 0; h < H; ++h) {
    float ex = __expf(eb[(size_t)e * H + h] - u2f_ord(mx[d * H + h]));
    eb[(size_t)e * H + h] = ex;
    atomicAdd(&den[d * H + h], ex);
  }
}

// phase 3 (bandwidth-dominant): out[dst] += h[src] * alpha ; one wave per edge,
// lanes cover the HC feature channels (coalesced 128B segments, L2-resident).
__global__ void edge_agg(const float* __restrict__ Hf,
                         const float* __restrict__ eb,
                         const float* __restrict__ den,
                         float* __restrict__ out, const int* __restrict__ ei,
                         int E, int Ep, int H, int C) {
  const int lane = threadIdx.x & 31;
  const int nwaves = (gridDim.x * blockDim.x) >> 5;
  const int HCd = H * C;
  for (int e = (blockIdx.x * blockDim.x + threadIdx.x) >> 5; e < Ep; e += nwaves) {
    int s, d;
    if (e < E) { s = ei[e]; d = ei[E + e]; } else { s = e - E; d = s; }
    const float* hrow = Hf + (size_t)s * HCd;
    float* orow = out + (size_t)d * HCd;
    __builtin_prefetch(hrow + lane, 0, 1);       // global_prefetch of gather row
    for (int c = lane; c < HCd; c += 32) {
      int h = c / C;
      float alpha = eb[(size_t)e * H + h] / (den[d * H + h] + 1e-16f);
      atomicAdd(&orow[c], hrow[c] * alpha);
    }
  }
}

// out = elu(out + bias); also emit f16 copy for the next layer's WMMA GEMM
__global__ void finalize_elu(float* __restrict__ out,
                             _Float16* __restrict__ out16,
                             const float* __restrict__ bias, int N, int HC) {
  int i = blockIdx.x * blockDim.x + threadIdx.x;
  if (i >= N * HC) return;
  float v = out[i] + bias[i % HC];
  v = (v > 0.0f) ? v : (__expf(v) - 1.0f);
  out[i] = v;
  out16[i] = (_Float16)v;
}

// ---- global mean pool ----
__global__ void pool_zero(float* __restrict__ g, float* __restrict__ cnt,
                          int G, int C) {
  int i = blockIdx.x * blockDim.x + threadIdx.x;
  if (i < G * C) g[i] = 0.0f;
  if (i < G) cnt[i] = 0.0f;
}

__global__ void pool_acc(const float* __restrict__ h,
                         const int* __restrict__ batch, float* __restrict__ g,
                         float* __restrict__ cnt, int N, int C) {
  int i = blockIdx.x * blockDim.x + threadIdx.x;
  if (i >= N * C) return;
  int n = i / C, c = i % C;
  int b = batch[n];
  atomicAdd(&g[(size_t)b * C + c], h[(size_t)n * C + c]);
  if (c == 0) atomicAdd(&cnt[b], 1.0f);
}

// ---- both MLP heads fused: out[0..G) = cls, out[G..2G) = reg ----
__global__ void heads(const float* __restrict__ g, const float* __restrict__ cnt,
                      const float* __restrict__ Wc1, const float* __restrict__ bc1,
                      const float* __restrict__ Wc2, const float* __restrict__ bc2,
                      const float* __restrict__ Wr1, const float* __restrict__ br1,
                      const float* __restrict__ Wr2, const float* __restrict__ br2,
                      float* __restrict__ out, int G, int C) {
  int b = blockIdx.x * blockDim.x + threadIdx.x;
  if (b >= G) return;
  float gv[64];
  float inv = 1.0f / fmaxf(cnt[b], 1.0f);
  for (int c = 0; c < C; ++c) gv[c] = g[(size_t)b * C + c] * inv;
  float accc = bc2[0], accr = br2[0];
  for (int j = 0; j < C; ++j) {
    float hc = bc1[j], hr = br1[j];
    const float* wc = Wc1 + (size_t)j * C;
    const float* wr = Wr1 + (size_t)j * C;
    for (int c = 0; c < C; ++c) { hc += gv[c] * wc[c]; hr += gv[c] * wr[c]; }
    accc += fmaxf(hc, 0.0f) * Wc2[j];
    accr += fmaxf(hr, 0.0f) * Wr2[j];
  }
  out[b] = accc;
  out[G + b] = accr;
}

// ============================================================================
extern "C" void kernel_launch(void* const* d_in, const int* in_sizes, int n_in,
                              void* d_out, int out_size, void* d_ws, size_t ws_size,
                              hipStream_t stream) {
  const float* x    = (const float*)d_in[0];
  const int*   ei   = (const int*)d_in[1];
  const int*   batch= (const int*)d_in[2];
  const float* W1 = (const float*)d_in[3],  *as1 = (const float*)d_in[4];
  const float* ad1= (const float*)d_in[5],  *b1  = (const float*)d_in[6];
  const float* W2 = (const float*)d_in[7],  *as2 = (const float*)d_in[8];
  const float* ad2= (const float*)d_in[9],  *b2  = (const float*)d_in[10];
  const float* W3 = (const float*)d_in[11], *as3 = (const float*)d_in[12];
  const float* ad3= (const float*)d_in[13], *b3  = (const float*)d_in[14];
  const float* Wc1= (const float*)d_in[15], *bc1 = (const float*)d_in[16];
  const float* Wc2= (const float*)d_in[17], *bc2 = (const float*)d_in[18];
  const float* Wr1= (const float*)d_in[19], *br1 = (const float*)d_in[20];
  const float* Wr2= (const float*)d_in[21], *br2 = (const float*)d_in[22];

  const int N   = in_sizes[2];
  const int E   = in_sizes[1] / 2;
  const int Ep  = E + N;                 // with self-loops
  const int IN  = in_sizes[0] / N;       // 128
  const int HC  = in_sizes[3] / IN;      // 256
  const int H1  = 4;                     // HEADS (reference constant)
  const int C1  = HC / H1;               // 64
  const int HID = in_sizes[11] / HC;     // 64
  const int G   = out_size / 2;          // 256 graphs

  // workspace carve-up (all region sizes are multiples of 16B)
  float*    bufA = (float*)d_ws;                        // [N, HC] WMMA features
  float*    bufB = bufA + (size_t)N * HC;               // [N, HC] aggregation out
  float*    aloB = bufB + (size_t)N * HC;               // [N, H]
  float*    ahiB = aloB + (size_t)N * H1;               // [N, H]
  unsigned* mxB  = (unsigned*)(ahiB + (size_t)N * H1);  // [N, H] ordered-uint max
  float*    denB = (float*)(mxB + (size_t)N * H1);      // [N, H]
  float*    ebB  = denB + (size_t)N * H1;               // [Ep, H] e then ex
  float*    gB   = ebB + (size_t)Ep * H1;               // [G, HID]
  float*    cntB = gB + (size_t)G * HID;                // [G]
  _Float16* Xh   = (_Float16*)(cntB + G);               // [N, IN]  f16 input
  _Float16* F16  = Xh + (size_t)N * IN;                 // [N, HC]  f16 layer out
  _Float16* Wh1  = F16 + (size_t)N * HC;                // [HC, IN]
  _Float16* Wh2  = Wh1 + (size_t)HC * IN;               // [HC, HC]
  _Float16* Wh3  = Wh2 + (size_t)HC * HC;               // [HID, HC]

  dim3 blk(256);
  auto cdiv = [](long long a, long long b) { return (unsigned)((a + b - 1) / b); };

  // one-time f16 conversions
  cvt_f32_f16<<<cdiv((long long)N * IN, 256), blk, 0, stream>>>(x, Xh, N * IN);
  cvt_f32_f16<<<cdiv((long long)HC * IN, 256), blk, 0, stream>>>(W1, Wh1, HC * IN);
  cvt_f32_f16<<<cdiv((long long)HC * HC, 256), blk, 0, stream>>>(W2, Wh2, HC * HC);
  cvt_f32_f16<<<cdiv((long long)HID * HC, 256), blk, 0, stream>>>(W3, Wh3, HID * HC);

  auto run_layer = [&](const _Float16* inF16, const _Float16* W16,
                       const float* asv, const float* adv, const float* bias,
                       int K, int Hh, int Cc, float* hf, float* outb) {
    const int HCl = Hh * Cc;
    dim3 gg(cdiv(N, 128), HCl / 64);
    gemm_xwt_wmma<<<gg, blk, 0, stream>>>(inF16, W16, hf, N, K, HCl);
    attn_coef<<<cdiv((long long)N * Hh, 256), blk, 0, stream>>>(hf, asv, adv, aloB, ahiB, N, Hh, Cc);
    layer_init<<<cdiv((long long)N * HCl, 256), blk, 0, stream>>>(mxB, denB, outb, N * Hh, N * HCl);
    edge_max<<<cdiv(Ep, 256), blk, 0, stream>>>(ei, aloB, ahiB, ebB, mxB, E, Ep, Hh);
    edge_exp<<<cdiv(Ep, 256), blk, 0, stream>>>(ei, ebB, mxB, denB, E, Ep, Hh);
    edge_agg<<<cdiv(Ep, 8), blk, 0, stream>>>(hf, ebB, denB, outb, ei, E, Ep, Hh, Cc);
    finalize_elu<<<cdiv((long long)N * HCl, 256), blk, 0, stream>>>(outb, F16, bias, N, HCl);
  };

  run_layer(Xh,  Wh1, as1, ad1, b1, IN, H1, C1,  bufA, bufB);
  run_layer(F16, Wh2, as2, ad2, b2, HC, H1, C1,  bufA, bufB);
  run_layer(F16, Wh3, as3, ad3, b3, HC, 1,  HID, bufA, bufB);

  pool_zero<<<cdiv((long long)G * HID, 256), blk, 0, stream>>>(gB, cntB, G, HID);
  pool_acc<<<cdiv((long long)N * HID, 256), blk, 0, stream>>>(bufB, batch, gB, cntB, N, HID);
  heads<<<cdiv(G, 256), blk, 0, stream>>>(gB, cntB, Wc1, bc1, Wc2, bc2, Wr1, br1,
                                          Wr2, br2, (float*)d_out, G, HID);
}
